// Encoder_47656957116696
// MI455X (gfx1250) — matrix operand order repaired and checked by reading, not compiled, not run on hardware
//
#include <hip/hip_runtime.h>

#define B_   16
#define T_   2048
#define DIN_ 64
#define F_   256
#define U_   256
#define G3_  768   // 3*U

typedef __attribute__((ext_vector_type(16))) _Float16 v16h;
typedef __attribute__((ext_vector_type(8)))  _Float16 v8h;
typedef __attribute__((ext_vector_type(8)))  float    v8f;

union F4 { float4 v; float f[4]; };

// ---- WMMA helpers (CDNA5 layouts, ISA 7.12.2) ----------------------------

// B-matrix 32x16 f16 tile fragment: per lane, 16 contiguous halves (32B)
// from a [N][K] (transposed, K contiguous) weight image:
//   n = tile*16 + (lane&15), K = k0*32 + (lane>>4)*16 + h
__device__ __forceinline__ v16h ld_b_frag(const _Float16* p) {
  v8h lo = *(const v8h*)(p);
  v8h hi = *(const v8h*)(p + 8);
  return __builtin_shufflevector(lo, hi, 0,1,2,3,4,5,6,7,8,9,10,11,12,13,14,15);
}

// A-matrix 16x32 f16 fragment from a row-major row pointer:
//   m = lane&15, halves[0..7] = row[kbase .. +7], halves[8..15] = row[kbase+8 .. +15]
//   where kbase = k0*32 + 16*(lane>>4)
__device__ __forceinline__ v16h ld_a_frag(const _Float16* row, int kbase) {
  v8h lo = *(const v8h*)(row + kbase);
  v8h hi = *(const v8h*)(row + kbase + 8);
  return __builtin_shufflevector(lo, hi, 0,1,2,3,4,5,6,7,8,9,10,11,12,13,14,15);
}

__device__ __forceinline__ v8f wmma_f16(v16h a, v16h b, v8f c) {
  return __builtin_amdgcn_wmma_f32_16x16x32_f16(false, a, false, b, (short)0, c,
                                                false, false);
}

__device__ __forceinline__ float sigmoidf_(float x) {
  return 1.0f / (1.0f + __expf(-x));
}

// ---- Kernel 0: cast + transpose weights to f16 [N=768][K=256] ------------
__global__ void cast_tr_kernel(const float* __restrict__ w0, const float* __restrict__ u0,
                               const float* __restrict__ w1, const float* __restrict__ u1,
                               _Float16* __restrict__ w0t, _Float16* __restrict__ u0t,
                               _Float16* __restrict__ w1t, _Float16* __restrict__ u1t) {
  int id = blockIdx.x * blockDim.x + threadIdx.x;
  if (id >= G3_ * F_) return;
  int n = id / F_;
  int k = id % F_;
  int src = k * G3_ + n;
  w0t[id] = (_Float16)w0[src];
  u0t[id] = (_Float16)u0[src];
  w1t[id] = (_Float16)w1[src];
  u1t[id] = (_Float16)u1[src];
}

// ---- Kernel 1: Conv1D(k=3, same) + ELU, f16 output -----------------------
__global__ void conv_elu_kernel(const float* __restrict__ x,   // [B*T, 64]
                                const float* __restrict__ w,   // [3, 64, 256]
                                const float* __restrict__ bias,// [256]
                                _Float16* __restrict__ out) {  // [B*T, 256]
  __shared__ float xs[3 * DIN_];
  const long row = blockIdx.x;            // b*T + t
  const int  t   = (int)(row & (T_ - 1));
  const int  f   = threadIdx.x;           // 0..255
  if (f < 3 * DIN_) {
    int tap = f >> 6, c = f & 63;
    int tt = t + tap - 1;
    float v = 0.0f;
    if (tt >= 0 && tt < T_) v = x[(row + tap - 1) * DIN_ + c];
    xs[f] = v;
  }
  __syncthreads();
  float acc = bias[f];
#pragma unroll 8
  for (int i = 0; i < 3 * DIN_; ++i)
    acc = fmaf(xs[i], w[i * F_ + f], acc);
  float e = acc > 0.0f ? acc : (__expf(acc) - 1.0f);
  out[row * F_ + f] = (_Float16)e;
}

// ---- Kernel 2: input-projection GEMM: C[M,768] = A[M,256] @ W^T + bi -----
// A: f16 row-major.  Bt: f16 [768][256] (N-major).  C: f32.
__global__ __launch_bounds__(256)
void xproj_gemm_kernel(const _Float16* __restrict__ A,
                       const _Float16* __restrict__ Bt,
                       const float* __restrict__ bias,   // bi = b[0], length 768
                       float* __restrict__ C) {
  const int tid  = threadIdx.x;
  const int lane = tid & 31;
  const int wave = tid >> 5;              // 0..7
  const int nn   = lane & 15;
  const int hi   = lane >> 4;             // 0/1
  const int m0   = blockIdx.x * 64 + (wave & 3) * 16;
  const int n0   = blockIdx.y * 128 + (wave >> 2) * 64;

  v8f acc[4] = {{}, {}, {}, {}};
  const _Float16* arow = A + (long)(m0 + nn) * F_;

#pragma unroll
  for (int k = 0; k < 8; ++k) {
    v16h a = ld_a_frag(arow, k * 32 + hi * 16);
#pragma unroll
    for (int j = 0; j < 4; ++j) {
      v16h b = ld_b_frag(Bt + (long)(n0 + j * 16 + nn) * F_ + k * 32 + hi * 16);
      acc[j] = wmma_f16(a, b, acc[j]);
    }
  }
#pragma unroll
  for (int j = 0; j < 4; ++j) {
    int n = n0 + j * 16 + nn;
    float bi = bias[n];
#pragma unroll
    for (int r = 0; r < 8; ++r)
      C[(long)(m0 + r + hi * 8) * G3_ + n] = acc[j][r] + bi;
  }
}

// ---- Kernel 3: GRU recurrent scan (single WGP, Ur resident in VGPRs) -----
// xp:  f32 [B*T, 768] (includes bi)
// urt: f16 [768][256] (N-major transposed Ur)
// bvec: f32 [2,768]; br = bvec+768
__global__ __launch_bounds__(768, 1)
void gru_scan_kernel(const float* __restrict__ xp,
                     const _Float16* __restrict__ urt,
                     const float* __restrict__ bvec,
                     _Float16* __restrict__ ys16,   // [B*T,256] or null
                     float* __restrict__ xout,      // [B*T,256] or null
                     float* __restrict__ states,    // [B,256,2]
                     int layer) {
  __shared__ _Float16 h_lds[B_][U_ + 8];     // padded rows, 16B-aligned stride
  __shared__ float    rp_lds[B_][G3_];

  const int tid  = threadIdx.x;              // 0..767 (24 waves)
  const int lane = tid & 31;
  const int wave = tid >> 5;                 // 0..23
  const int nn   = lane & 15;
  const int hi   = lane >> 4;
  const float* br = bvec + G3_;

  // zero-init hidden state
  for (int i = tid; i < B_ * (U_ + 8); i += 768) ((_Float16*)h_lds)[i] = (_Float16)0.0f;

  // Preload Ur B-fragments for this wave's two N-tiles (loop-invariant: stays in VGPRs)
  const int tile0 = wave * 2;
  const int tile1 = wave * 2 + 1;
  v16h bf0[8], bf1[8];
  {
    const _Float16* p0 = urt + (long)(tile0 * 16 + nn) * F_ + hi * 16;
    const _Float16* p1 = urt + (long)(tile1 * 16 + nn) * F_ + hi * 16;
#pragma unroll
    for (int k = 0; k < 8; ++k) {
      bf0[k] = ld_b_frag(p0 + k * 32);
      bf1[k] = ld_b_frag(p1 + k * 32);
    }
  }
  __syncthreads();

  for (int t = 0; t < T_; ++t) {
    // ---- issue this step's xp loads early (overlap with WMMA below) ----
    F4 xz[2], xr[2], xh[2];
    int gidx[2], ng = 0;
    for (int g = tid; g < (B_ * U_) / 4; g += 768) {   // 1024 groups of 4
      int m  = g >> 6;
      int u0 = (g & 63) * 4;
      const float* xrow = xp + ((long)m * T_ + t) * G3_;
      xz[ng].v = *(const float4*)(xrow + u0);
      xr[ng].v = *(const float4*)(xrow + U_ + u0);
      xh[ng].v = *(const float4*)(xrow + 2 * U_ + u0);
      if (t + 8 < T_) {
        const float* pf = xp + ((long)m * T_ + t + 8) * G3_;
        __builtin_prefetch(pf + u0, 0, 1);
        __builtin_prefetch(pf + U_ + u0, 0, 1);
        __builtin_prefetch(pf + 2 * U_ + u0, 0, 1);
      }
      gidx[ng] = g;
      ++ng;
    }

    // ---- phase 1: rp = h @ Ur via WMMA (h from LDS, Ur from registers) ----
    v8f acc0 = {}, acc1 = {};
    const _Float16* hrow = &h_lds[nn][0];   // A-frag row: m = lane&15
#pragma unroll
    for (int k = 0; k < 8; ++k) {
      v16h a = ld_a_frag(hrow, k * 32 + hi * 16);
      acc0 = wmma_f16(a, bf0[k], acc0);
      acc1 = wmma_f16(a, bf1[k], acc1);
    }
#pragma unroll
    for (int r = 0; r < 8; ++r) {
      rp_lds[r + hi * 8][tile0 * 16 + nn] = acc0[r];
      rp_lds[r + hi * 8][tile1 * 16 + nn] = acc1[r];
    }
    __syncthreads();

    // ---- phase 2: gate math + h update ----
    for (int i = 0; i < ng; ++i) {
      int g  = gidx[i];
      int m  = g >> 6;
      int u0 = (g & 63) * 4;
      long rowi = (long)m * T_ + t;
#pragma unroll
      for (int j = 0; j < 4; ++j) {
        int u = u0 + j;
        float hz = rp_lds[m][u]          + br[u];
        float hr = rp_lds[m][U_ + u]     + br[U_ + u];
        float hh = rp_lds[m][2 * U_ + u] + br[2 * U_ + u];
        float hprev = (float)h_lds[m][u];
        float z  = sigmoidf_(xz[i].f[j] + hz);
        float rr = sigmoidf_(xr[i].f[j] + hr);
        float hc = tanhf(xh[i].f[j] + rr * hh);
        float hn = z * hprev + (1.0f - z) * hc;
        h_lds[m][u] = (_Float16)hn;
        if (ys16) ys16[rowi * U_ + u] = (_Float16)hn;
        if (xout) xout[rowi * U_ + u] = hn;
        if (t == T_ - 1) states[(m * U_ + u) * 2 + layer] = hn;
      }
    }
    __syncthreads();
  }
}

// ---- Host-side launcher ---------------------------------------------------
extern "C" void kernel_launch(void* const* d_in, const int* in_sizes, int n_in,
                              void* d_out, int out_size, void* d_ws, size_t ws_size,
                              hipStream_t stream) {
  (void)in_sizes; (void)n_in; (void)out_size; (void)ws_size;
  const float* input_seq = (const float*)d_in[0];
  const float* conv_w    = (const float*)d_in[1];
  const float* conv_b    = (const float*)d_in[2];
  const float* w0        = (const float*)d_in[3];
  const float* u0        = (const float*)d_in[4];
  const float* b0        = (const float*)d_in[5];
  const float* w1        = (const float*)d_in[6];
  const float* u1        = (const float*)d_in[7];
  const float* b1        = (const float*)d_in[8];

  float* xout   = (float*)d_out;                              // [B,T,256]
  float* states = xout + (size_t)B_ * T_ * U_;                // [B,256,2]

  char* ws = (char*)d_ws;
  _Float16* conv16 = (_Float16*)ws; ws += (size_t)B_ * T_ * F_ * 2;      // 16 MB
  _Float16* ys0    = (_Float16*)ws; ws += (size_t)B_ * T_ * U_ * 2;      // 16 MB
  float*    xpbuf  = (float*)ws;    ws += (size_t)B_ * T_ * G3_ * 4;     // 96 MB
  _Float16* w0t    = (_Float16*)ws; ws += (size_t)G3_ * F_ * 2;
  _Float16* u0t    = (_Float16*)ws; ws += (size_t)G3_ * F_ * 2;
  _Float16* w1t    = (_Float16*)ws; ws += (size_t)G3_ * F_ * 2;
  _Float16* u1t    = (_Float16*)ws; ws += (size_t)G3_ * F_ * 2;

  cast_tr_kernel<<<(G3_ * F_ + 255) / 256, 256, 0, stream>>>(
      w0, u0, w1, u1, w0t, u0t, w1t, u1t);

  conv_elu_kernel<<<B_ * T_, F_, 0, stream>>>(input_seq, conv_w, conv_b, conv16);

  // layer 0
  xproj_gemm_kernel<<<dim3((B_ * T_) / 64, G3_ / 128), 256, 0, stream>>>(
      conv16, w0t, b0, xpbuf);
  gru_scan_kernel<<<1, 768, 0, stream>>>(xpbuf, u0t, b0, ys0, nullptr, states, 0);

  // layer 1
  xproj_gemm_kernel<<<dim3((B_ * T_) / 64, G3_ / 128), 256, 0, stream>>>(
      ys0, w1t, b1, xpbuf);
  gru_scan_kernel<<<1, 768, 0, stream>>>(xpbuf, u1t, b1, nullptr, xout, states, 1);
}
